// CorrespodenceNet_14877766713953
// MI455X (gfx1250) — compile-verified, matrix operand order checked by reading, not run on hardware
//
#include <hip/hip_runtime.h>

// ---------------------------------------------------------------------------
// MI455X fused correspondence kernel:
//   corr = norm(X) @ norm(Y)^T  (9216x9216, K=256)  -- never materialized
//   softmax(corr/0.01) @ ref_ab   +  rowmax(corr)
// v_wmma_f32_16x16x32_f16 with online softmax.  Key tiles staged into LDS in
// fragment-swizzled layout via GLOBAL_LOAD_ASYNC_TO_LDS_B128 (ASYNCcnt), and
// the softmax update is batched over 8 key tiles (128 keys) to amortize the
// butterfly shuffle reductions (shuffles/WMMA = 2:1).
// Features pre-scaled by 10 so the WMMA output == corr/TAU directly.
// ---------------------------------------------------------------------------

#define HWPIX   9216
#define CCH     256
#define NQT     576          // 9216/16 query tiles
#define KSPLIT  4            // key-dimension split for occupancy
#define KT_PER  144          // 576/KSPLIT key tiles per split
#define NT      4            // key tiles per LDS stage (32KB)
#define NB      8            // key tiles per softmax update (2 stages, 128 keys)
#define NSTAGE  (KT_PER/NT)  // 36 stages per block (even -> pairs)
#define WPB     8            // waves per block (wave32) -> 256 threads
#define STAGE_H (NT*16*CCH)  // halves per staged group (16384 = 32KB)
#define STAGE_B (STAGE_H*2)  // bytes per staged group

typedef __attribute__((ext_vector_type(16))) _Float16 v16h;
typedef __attribute__((ext_vector_type(8)))  _Float16 v8h;
typedef __attribute__((ext_vector_type(8)))  float    v8f;

// -------------------------- channel statistics -----------------------------

__global__ void cn_init_stats(float* s) {
    s[blockIdx.x * 256 + threadIdx.x] = 0.0f;   // grid 4 x 256 -> 1024 floats
}

__global__ void cn_stats(const float* __restrict__ x, const float* __restrict__ y,
                         float* xsum, float* xsq, float* ysum, float* ysq) {
    const int c  = threadIdx.x;          // channel (coalesced across lanes)
    const int r0 = blockIdx.x * 256;     // 36 blocks x 256 rows
    float sx = 0.f, sx2 = 0.f, sy = 0.f, sy2 = 0.f;
    for (int rr = 0; rr < 256; ++rr) {
        float v = x[(size_t)(r0 + rr) * CCH + c];
        sx += v; sx2 += v * v;
        float w = y[(size_t)(r0 + rr) * CCH + c];
        sy += w; sy2 += w * w;
    }
    atomicAdd(&xsum[c], sx);  atomicAdd(&xsq[c], sx2);
    atomicAdd(&ysum[c], sy);  atomicAdd(&ysq[c], sy2);
}

// normalize + convert to f16, pre-scaled by 10 (so f16 dot == corr/TAU)
__global__ void cn_normalize(const float* __restrict__ x, const float* __restrict__ y,
                             const float* __restrict__ xsum, const float* __restrict__ xsq,
                             const float* __restrict__ ysum, const float* __restrict__ ysq,
                             _Float16* __restrict__ xn, _Float16* __restrict__ yn) {
    const int i = blockIdx.x * blockDim.x + threadIdx.x;   // grid 9216 x 256 == exact
    const int c = i & (CCH - 1);
    const float mx = xsum[c] * (1.0f / HWPIX);
    const float rx = __frsqrt_rn(xsq[c]) * 10.0f;
    const float my = ysum[c] * (1.0f / HWPIX);
    const float ry = __frsqrt_rn(ysq[c]) * 10.0f;
    xn[i] = (_Float16)((x[i] - mx) * rx);
    yn[i] = (_Float16)((y[i] - my) * ry);
}

// ------------------------- fused GEMM + softmax ----------------------------

// A fragment from row-major global: lane holds row (lane&15); halves are
// K = kbase+off0+{0..7} and kbase+off0+16+{0..7}, off0 = (lane>=16)?8:0.
__device__ __forceinline__ v16h cn_load_frag_g(const _Float16* p) {
    v8h lo = *(const v8h*)(p);
    v8h hi = *(const v8h*)(p + 16);
    return __builtin_shufflevector(lo, hi, 0, 1, 2, 3, 4, 5, 6, 7,
                                           8, 9, 10, 11, 12, 13, 14, 15);
}

// B fragment from swizzled LDS: each lane's 16 halves are contiguous (32B)
__device__ __forceinline__ v16h cn_load_frag_l(const _Float16* p) {
    v8h lo = *(const v8h*)(p);
    v8h hi = *(const v8h*)(p + 8);
    return __builtin_shufflevector(lo, hi, 0, 1, 2, 3, 4, 5, 6, 7,
                                           8, 9, 10, 11, 12, 13, 14, 15);
}

// Async-stage NT=4 key tiles (32KB) from row-major global into fragment-
// swizzled LDS (dst[tile][kk][lane][16 halves]) with per-lane LDS scatter:
// GLOBAL_LOAD_ASYNC_TO_LDS_B128 takes a global addr VGPR pair and an LDS
// destination address VGPR; no data VGPRs, tracked by ASYNCcnt.
// Chunk v (16B = 8 halves): tile = v>>9, n = (v>>5)&15, kk = (v>>2)&7,
// o = (v&3)*8; lane = n + 16*(o&8?1:0); second 16B half if o>=16.
__device__ __forceinline__ void cn_stage_async(unsigned int lds_base,
                                               const _Float16* src, int tid) {
    const unsigned long long gbase = (unsigned long long)(uintptr_t)src;
#pragma unroll
    for (int j = 0; j < 8; ++j) {
        int v    = tid + j * 256;                 // 0..2047 chunks
        int tile = v >> 9;
        int n    = (v >> 5) & 15;
        int kk   = (v >> 2) & 7;
        int lane = n + ((v & 1) << 4);
        int hsel = (v & 2) ? 16 : 0;              // bytes
        unsigned int dst = lds_base + tile * 8192 + kk * 1024 + lane * 32 + hsel;
        unsigned long long ga = gbase + (unsigned int)v * 16u;
        asm volatile("global_load_async_to_lds_b128 %0, %1, off"
                     :: "v"(dst), "v"(ga) : "memory");
    }
}

__device__ __forceinline__ void cn_wait_async() {
    asm volatile("s_wait_asynccnt 0x0" ::: "memory");
}

__global__ __launch_bounds__(WPB * 32) void cn_corr_softmax(
        const _Float16* __restrict__ xn, const _Float16* __restrict__ yn,
        const float* __restrict__ ref,
        float* __restrict__ pM, float* __restrict__ pL,
        float* __restrict__ pA, float* __restrict__ pB) {
    __shared__ _Float16 smem[2][STAGE_H];         // 2 x 32KB = 64KB

    const int tid    = threadIdx.x;
    const int wave   = tid >> 5;
    const int lane   = tid & 31;
    const int half16 = lane & 15;
    const int off0   = (lane >= 16) ? 8 : 0;
    const int mtile  = blockIdx.x * WPB + wave;
    const int kbase  = blockIdx.y * KT_PER;
    const unsigned int lds0 = (unsigned int)(uintptr_t)&smem[0][0];

    // A panel: this wave's 16 query rows, full K=256, kept in 64 VGPRs/lane
    const _Float16* arow = xn + (size_t)(mtile * 16 + half16) * CCH;
    v16h afrag[8];
#pragma unroll
    for (int kk = 0; kk < 8; ++kk)
        afrag[kk] = cn_load_frag_g(arow + kk * 32 + off0);

    // online-softmax state, one slot per D-matrix VGPR row (replicated x16 lanes)
    float mr[8], lr[8], aA[8], aB[8];
#pragma unroll
    for (int r = 0; r < 8; ++r) { mr[r] = -3.0e38f; lr[r] = 0.f; aA[r] = 0.f; aB[r] = 0.f; }

    cn_stage_async(lds0, yn + (size_t)kbase * 16 * CCH, tid);   // prefetch stage 0

    for (int gg = 0; gg < NSTAGE; gg += 2) {
        v8f acc[NB];
#pragma unroll
        for (int h = 0; h < 2; ++h) {
            const int g = gg + h;
            cn_wait_async();                      // our async writes to buf[g&1] done
            __syncthreads();                      // buf[g&1] ready, buf[(g+1)&1] free
            if (g + 1 < NSTAGE)                   // async-prefetch next group
                cn_stage_async(lds0 + ((g + 1) & 1) * STAGE_B,
                               yn + (size_t)(kbase + (g + 1) * NT) * 16 * CCH, tid);

            const _Float16* base = &smem[g & 1][lane * 16];
#pragma unroll
            for (int t = 0; t < NT; ++t) {
                v8f a = {0.f, 0.f, 0.f, 0.f, 0.f, 0.f, 0.f, 0.f};
#pragma unroll
                for (int kk = 0; kk < 8; ++kk) {
                    v16h bfrag = cn_load_frag_l(base + t * 4096 + kk * 512);
                    a = __builtin_amdgcn_wmma_f32_16x16x32_f16(
                            false, afrag[kk], false, bfrag,
                            (short)0, a, false, false);
                }
                acc[h * NT + t] = a;
            }
        }

        // reference ab colors for this lane's 8 key columns (128 keys total)
        float ra[NB], rb[NB];
#pragma unroll
        for (int t = 0; t < NB; ++t) {
            int n = (kbase + gg * NT + t) * 16 + half16;
            ra[t] = ref[HWPIX + n];
            rb[t] = ref[2 * HWPIX + n];
        }

#pragma unroll
        for (int r = 0; r < 8; ++r) {
            // per-lane max over the 8 tiles, then one butterfly over 16 N lanes
            float tm = fmaxf(fmaxf(fmaxf(acc[0][r], acc[1][r]),
                                   fmaxf(acc[2][r], acc[3][r])),
                             fmaxf(fmaxf(acc[4][r], acc[5][r]),
                                   fmaxf(acc[6][r], acc[7][r])));
            tm = fmaxf(tm, __shfl_xor(tm, 1, 32));
            tm = fmaxf(tm, __shfl_xor(tm, 2, 32));
            tm = fmaxf(tm, __shfl_xor(tm, 4, 32));
            tm = fmaxf(tm, __shfl_xor(tm, 8, 32));
            float nm    = fmaxf(mr[r], tm);
            float alpha = __expf(mr[r] - nm);
            float ps = 0.f, pa = 0.f, pb = 0.f;
#pragma unroll
            for (int t = 0; t < NB; ++t) {
                float p = __expf(acc[t][r] - nm);
                ps += p;
                pa = fmaf(p, ra[t], pa);
                pb = fmaf(p, rb[t], pb);
            }
            ps += __shfl_xor(ps, 1, 32); pa += __shfl_xor(pa, 1, 32); pb += __shfl_xor(pb, 1, 32);
            ps += __shfl_xor(ps, 2, 32); pa += __shfl_xor(pa, 2, 32); pb += __shfl_xor(pb, 2, 32);
            ps += __shfl_xor(ps, 4, 32); pa += __shfl_xor(pa, 4, 32); pb += __shfl_xor(pb, 4, 32);
            ps += __shfl_xor(ps, 8, 32); pa += __shfl_xor(pa, 8, 32); pb += __shfl_xor(pb, 8, 32);
            lr[r] = lr[r] * alpha + ps;
            aA[r] = aA[r] * alpha + pa;
            aB[r] = aB[r] * alpha + pb;
            mr[r] = nm;
        }
    }

    // one representative lane per 16-lane N group writes 8 query rows
    if (half16 == 0) {
        const int mloc = (lane >= 16) ? 8 : 0;
#pragma unroll
        for (int r = 0; r < 8; ++r) {
            const int    q   = mtile * 16 + mloc + r;
            const size_t idx = (size_t)blockIdx.y * HWPIX + q;
            pM[idx] = mr[r]; pL[idx] = lr[r]; pA[idx] = aA[r]; pB[idx] = aB[r];
        }
    }
}

// ----------------------- key-split log-sum-exp merge -----------------------

__global__ void cn_merge(const float* __restrict__ pM, const float* __restrict__ pL,
                         const float* __restrict__ pA, const float* __restrict__ pB,
                         float* __restrict__ out) {
    const int q = blockIdx.x * blockDim.x + threadIdx.x;   // 36 x 256 == 9216
    float M = -3.0e38f;
#pragma unroll
    for (int s = 0; s < KSPLIT; ++s) M = fmaxf(M, pM[s * HWPIX + q]);
    float L = 0.f, A = 0.f, B = 0.f;
#pragma unroll
    for (int s = 0; s < KSPLIT; ++s) {
        float e = __expf(pM[s * HWPIX + q] - M);
        L += pL[s * HWPIX + q] * e;
        A += pA[s * HWPIX + q] * e;
        B += pB[s * HWPIX + q] * e;
    }
    const float inv = 1.0f / L;
    out[q]             = A * inv;      // W_out channel a  [1,2,96,96] plane 0
    out[HWPIX + q]     = B * inv;      // W_out channel b  plane 1
    out[2 * HWPIX + q] = M * 0.01f;    // S_out: max corr == (corr/TAU)*TAU
}

// ------------------------------- launcher ----------------------------------

extern "C" void kernel_launch(void* const* d_in, const int* in_sizes, int n_in,
                              void* d_out, int out_size, void* d_ws, size_t ws_size,
                              hipStream_t stream) {
    const float* x   = (const float*)d_in[0];   // [9216,256]
    const float* y   = (const float*)d_in[1];   // [9216,256]
    const float* ref = (const float*)d_in[2];   // [1,3,96,96]
    float* out = (float*)d_out;                 // 27648 floats

    char* ws = (char*)d_ws;
    float* xsum = (float*)ws;                   // 256
    float* xsq  = xsum + 256;
    float* ysum = xsq  + 256;
    float* ysq  = ysum + 256;                   // stats: 4 KB total
    float* pM = (float*)(ws + 4096);            // KSPLIT x 9216 partials
    float* pL = pM + (size_t)KSPLIT * HWPIX;
    float* pA = pL + (size_t)KSPLIT * HWPIX;
    float* pB = pA + (size_t)KSPLIT * HWPIX;
    _Float16* xn = (_Float16*)(ws + 4096 + (size_t)4 * KSPLIT * HWPIX * sizeof(float));
    _Float16* yn = xn + (size_t)HWPIX * CCH;    // ~9.6 MB workspace total

    cn_init_stats<<<4, 256, 0, stream>>>(xsum);
    cn_stats<<<HWPIX / 256, 256, 0, stream>>>(x, y, xsum, xsq, ysum, ysq);
    cn_normalize<<<HWPIX, 256, 0, stream>>>(x, y, xsum, xsq, ysum, ysq, xn, yn);

    dim3 grid(NQT / WPB, KSPLIT);               // (72, 4)
    cn_corr_softmax<<<grid, WPB * 32, 0, stream>>>(xn, yn, ref, pM, pL, pA, pB);

    cn_merge<<<HWPIX / 256, 256, 0, stream>>>(pM, pL, pA, pB, out);
}
